// FusedConvBN_56908316672710
// MI455X (gfx1250) — matrix operand order repaired and checked by reading, not compiled, not run on hardware
//
#include <hip/hip_runtime.h>
#include <math.h>

typedef __attribute__((ext_vector_type(16))) _Float16 v16h;
typedef __attribute__((ext_vector_type(8)))  float    v8f;

#define CIN    32
#define COUT   32
#define HH     256
#define WWD    256
#define BATCH  16
#define TILE_W 64
#define HALO_W 66
#define ROWS_PER_BLK 4
#define LDS_ROWS 6            // ROWS_PER_BLK + 2 halo rows
#define KSTEPS 9
#define MAXV   32767.0f
#define EPSBN  1e-5f
#define NPIX_TOT 1048576.0f   // 16*256*256 per channel

// ---- workspace layout (bytes) ----
#define WS_SUM      0                  // 32 floats
#define WS_SUMSQ    128                // 32 floats
#define WS_BIAS     256                // 32 floats (quantized folded bias)
#define WS_XMAX     384                // 1 uint (absmax(X) bits)
#define AFRAG_ELTS  (2*KSTEPS*32*16)   // 9216 halves per plane
#define AFRAG_BYTES (AFRAG_ELTS*2)     // 18432
#define WS_ARAW_HI  1024
#define WS_ARAW_LO  (WS_ARAW_HI + 1*AFRAG_BYTES)
#define WS_AFOLD_HI (WS_ARAW_HI + 2*AFRAG_BYTES)
#define WS_AFOLD_LO (WS_ARAW_HI + 3*AFRAG_BYTES)

__device__ inline void split_f16(float v, _Float16& hi, _Float16& lo) {
    hi = (_Float16)v;
    lo = (_Float16)(v - (float)hi);
}

// Decode A-fragment element index -> (flat OIHW weight index, cout).
// Fragment storage order: (((mt*9 + ks)*32 + lane)*16 + half)
// A k order chosen here (contiguous, B matches): lane L, half i -> K_local = i + 16*(L>=16)
// K ordering chosen as k = ks*32 + cin with ks = kh*3 + kw  (so K_local == cin)
__device__ inline void frag_decode(int i, int& flat, int& cout) {
    int h16  = i & 15;
    int lane = (i >> 4) & 31;
    int ks   = (i >> 9) % KSTEPS;
    int mt   = i / (16*32*KSTEPS);
    int cin  = h16 + ((lane >= 16) ? 16 : 0);
    int kh = ks / 3, kw = ks % 3;
    cout = mt*16 + (lane & 15);
    flat = ((cout*CIN + cin)*3 + kh)*3 + kw;
}

// ---------------- init: zero stats + permute raw weights into A fragments ----
__global__ __launch_bounds__(256) void qconv_init(
    const float* __restrict__ W, float* __restrict__ wsf,
    unsigned* __restrict__ wsu,
    _Float16* __restrict__ a_hi, _Float16* __restrict__ a_lo)
{
    int tid = threadIdx.x;
    if (tid < 32) { wsf[tid] = 0.f; wsf[32 + tid] = 0.f; }
    if (tid == 0) wsu[WS_XMAX/4] = 0u;
    for (int i = tid; i < AFRAG_ELTS; i += 256) {
        int flat, cout;
        frag_decode(i, flat, cout);
        float wv = W[flat];
        _Float16 hi, lo; split_f16(wv, hi, lo);
        a_hi[i] = hi; a_lo[i] = lo;
    }
}

// ---------------- fold: BN fold + quantize weight/bias + permute -------------
__global__ __launch_bounds__(256) void qconv_fold(
    const float* __restrict__ W, const float* __restrict__ gamma,
    const float* __restrict__ beta, float* __restrict__ wsf,
    _Float16* __restrict__ af_hi, _Float16* __restrict__ af_lo)
{
    __shared__ float s_a[32], s_b[32];
    __shared__ unsigned s_wmax, s_bmax;
    int tid = threadIdx.x;
    if (tid == 0) { s_wmax = 0u; s_bmax = 0u; }
    __syncthreads();
    if (tid < 32) {
        float sum = wsf[tid], ssq = wsf[32 + tid];
        float mean = sum / NPIX_TOT;
        float var  = (ssq - NPIX_TOT * mean * mean) / (NPIX_TOT - 1.0f);
        float inv  = gamma[tid] / sqrtf(var + EPSBN);
        s_a[tid] = inv;
        float b = beta[tid] - mean * inv;
        s_b[tid] = b;
        atomicMax(&s_bmax, __float_as_uint(fabsf(b)));
    }
    __syncthreads();
    float lm = 0.f;
    for (int i = tid; i < COUT*CIN*9; i += 256) {
        float wv = s_a[i / (CIN*9)] * W[i];
        lm = fmaxf(lm, fabsf(wv));
    }
    atomicMax(&s_wmax, __float_as_uint(lm));
    __syncthreads();
    float wmax = __uint_as_float(s_wmax); if (wmax == 0.f) wmax = 1.f;
    float sw = MAXV / wmax, rw = wmax / MAXV;
    for (int i = tid; i < AFRAG_ELTS; i += 256) {
        int flat, cout;
        frag_decode(i, flat, cout);
        float wv = s_a[cout] * W[flat];
        float wq = fminf(fmaxf(rintf(sw * wv), -MAXV), MAXV) * rw;
        _Float16 hi, lo; split_f16(wq, hi, lo);
        af_hi[i] = hi; af_lo[i] = lo;
    }
    if (tid < 32) {
        float bmax = __uint_as_float(s_bmax); if (bmax == 0.f) bmax = 1.f;
        float sb = MAXV / bmax, rb = bmax / MAXV;
        float bq = fminf(fmaxf(rintf(sb * s_b[tid]), -MAXV), MAXV) * rb;
        wsf[WS_BIAS/4 + tid] = bq;
    }
}

// ---------------- conv via implicit GEMM + WMMA f16 (hi/lo split) ------------
// Block = 256 threads = 8 waves. Block tile: 32 couts x 4 rows x 64 pixels.
// Wave w: mtile = w/4 (cout 16-chunk), c0 = 16*(w%4) (column subtile),
// each wave owns 4 row-tiles (4 independent accumulator chains).
template <bool QUANT>
__global__ __launch_bounds__(256) void qconv_wmma(
    const float* __restrict__ X,
    const _Float16* __restrict__ a_hi, const _Float16* __restrict__ a_lo,
    float* __restrict__ wsf, unsigned* __restrict__ wsu,
    float* __restrict__ out)
{
    __shared__ __attribute__((aligned(32))) _Float16 sXhi[LDS_ROWS*HALO_W*CIN];
    __shared__ __attribute__((aligned(32))) _Float16 sXlo[LDS_ROWS*HALO_W*CIN];
    __shared__ float s_sum[32], s_sumsq[32];
    __shared__ unsigned s_xmax;

    int tid = threadIdx.x;
    int gid = blockIdx.x;
    int n   = gid >> 8;                 // 256 blocks per image (64 rowgrps * 4)
    int rem = gid & 255;
    int h0  = (rem >> 2) << 2;          // 4 output rows per block
    int w0  = (rem & 3) * TILE_W;

    if (!QUANT) {
        if (tid < 32) { s_sum[tid] = 0.f; s_sumsq[tid] = 0.f; }
        if (tid == 0) s_xmax = 0u;
    }

    float sx = 1.f, rx = 1.f;
    if (QUANT) {
        float xmax = __uint_as_float(wsu[WS_XMAX/4]);
        if (xmax == 0.f) xmax = 1.f;
        sx = MAXV / xmax; rx = xmax / MAXV;
    }

    // Stage halo tile X[n, :, h0-1..h0+4, w0-1..w0+64] -> LDS [row][col][cin]
    float lmax = 0.f;
    for (int i = tid; i < LDS_ROWS*CIN*HALO_W; i += 256) {
        int col = i % HALO_W;
        int t   = i / HALO_W;
        int cin = t & 31;
        int row = t >> 5;               // 0..5
        int hh = h0 + row - 1;
        int ww = w0 + col - 1;
        float v = 0.f;
        if (hh >= 0 && hh < HH && ww >= 0 && ww < WWD)
            v = X[((n*CIN + cin)*HH + hh)*WWD + ww];
        if (!QUANT) lmax = fmaxf(lmax, fabsf(v));
        if (QUANT)  v = fminf(fmaxf(rintf(sx * v), -MAXV), MAXV) * rx;
        _Float16 hi, lo; split_f16(v, hi, lo);
        int di = (row*HALO_W + col)*CIN + cin;
        sXhi[di] = hi; sXlo[di] = lo;
    }
    __syncthreads();
    if (!QUANT) atomicMax(&s_xmax, __float_as_uint(lmax));

    int lane  = tid & 31;
    int wave  = tid >> 5;
    int mt    = wave >> 2;
    int c0    = (wave & 3) << 4;
    int laneN = lane & 15;
    int cin0  = (lane >> 4) << 4;       // B frag: lanes 0-15 -> K 0-15, 16-31 -> K 16-31

    const v16h* Ah = (const v16h*)a_hi;
    const v16h* Al = (const v16h*)a_lo;
    v8f acc[ROWS_PER_BLK] = {};
#pragma unroll
    for (int ks = 0; ks < KSTEPS; ++ks) {
        int kh = ks / 3, kw = ks % 3;
        v16h ah = Ah[(mt*KSTEPS + ks)*32 + lane];
        v16h al = Al[(mt*KSTEPS + ks)*32 + lane];
        int colb = c0 + laneN + kw;
#pragma unroll
        for (int row = 0; row < ROWS_PER_BLK; ++row) {
            int bidx = (((row + kh)*HALO_W + colb)*CIN + cin0);
            v16h bh = *(const v16h*)&sXhi[bidx];
            v16h bl = *(const v16h*)&sXlo[bidx];
            acc[row] = __builtin_amdgcn_wmma_f32_16x16x32_f16(false, ah, false, bh, (short)0, acc[row], false, false);
            acc[row] = __builtin_amdgcn_wmma_f32_16x16x32_f16(false, ah, false, bl, (short)0, acc[row], false, false);
            acc[row] = __builtin_amdgcn_wmma_f32_16x16x32_f16(false, al, false, bh, (short)0, acc[row], false, false);
        }
    }

    if (!QUANT) {
        // D layout: vgpr r, lane L -> cout = mt*16 + r + 8*(L/16), pix = L%16
#pragma unroll
        for (int r = 0; r < 8; ++r) {
            float v0 = acc[0][r], v1 = acc[1][r], v2 = acc[2][r], v3 = acc[3][r];
            float s = (v0 + v1) + (v2 + v3);
            float q = ((v0*v0 + v1*v1) + (v2*v2 + v3*v3));
#pragma unroll
            for (int off = 1; off < 16; off <<= 1) {
                s += __shfl_xor(s, off, 32);
                q += __shfl_xor(q, off, 32);
            }
            if (laneN == 0) {
                int c = mt*16 + r + ((lane >> 4) << 3);
                atomicAdd(&s_sum[c], s);
                atomicAdd(&s_sumsq[c], q);
            }
        }
        __syncthreads();
        if (tid < 32) {
            atomicAdd(&wsf[tid],      s_sum[tid]);
            atomicAdd(&wsf[32 + tid], s_sumsq[tid]);
        }
        if (tid == 0) atomicMax(&wsu[WS_XMAX/4], s_xmax);
    } else {
        const float* biasq = wsf + WS_BIAS/4;
#pragma unroll
        for (int r = 0; r < 8; ++r) {
            int c  = mt*16 + r + ((lane >> 4) << 3);
            int ww = w0 + c0 + laneN;
            float b = biasq[c];
#pragma unroll
            for (int row = 0; row < ROWS_PER_BLK; ++row) {
                out[((n*COUT + c)*HH + (h0 + row))*WWD + ww] = acc[row][r] + b;
            }
        }
    }
}

extern "C" void kernel_launch(void* const* d_in, const int* in_sizes, int n_in,
                              void* d_out, int out_size, void* d_ws, size_t ws_size,
                              hipStream_t stream) {
    const float* X     = (const float*)d_in[0];
    const float* W     = (const float*)d_in[1];
    const float* gamma = (const float*)d_in[2];
    const float* beta  = (const float*)d_in[3];
    // d_in[4]/d_in[5] (running stats) unused by the training-mode reference.
    float* out = (float*)d_out;

    char* ws = (char*)d_ws;
    float*    wsf      = (float*)ws;
    unsigned* wsu      = (unsigned*)ws;
    _Float16* araw_hi  = (_Float16*)(ws + WS_ARAW_HI);
    _Float16* araw_lo  = (_Float16*)(ws + WS_ARAW_LO);
    _Float16* afold_hi = (_Float16*)(ws + WS_AFOLD_HI);
    _Float16* afold_lo = (_Float16*)(ws + WS_AFOLD_LO);

    const int nblk = BATCH * (HH / ROWS_PER_BLK) * (WWD / TILE_W);   // 4096

    qconv_init<<<1, 256, 0, stream>>>(W, wsf, wsu, araw_hi, araw_lo);
    qconv_wmma<false><<<nblk, 256, 0, stream>>>(X, araw_hi, araw_lo, wsf, wsu, nullptr);
    qconv_fold<<<1, 256, 0, stream>>>(W, gamma, beta, wsf, afold_hi, afold_lo);
    qconv_wmma<true><<<nblk, 256, 0, stream>>>(X, afold_hi, afold_lo, wsf, wsu, out);
}
